// MultiHeadAttention_5703716569110
// MI455X (gfx1250) — matrix operand order repaired and checked by reading, not compiled, not run on hardware
//
#include <hip/hip_runtime.h>
#include <hip/hip_bf16.h>

// ---------------------------------------------------------------------------
// MI455X (gfx1250) multi-head attention forward, bf16 WMMA / f32 accumulate.
// GEMM: 64x64 register tile per wave (16 WMMAs / K-step) -> 32 FLOP/byte.
// ---------------------------------------------------------------------------

typedef __attribute__((ext_vector_type(16))) __bf16 v16bf;
typedef __attribute__((ext_vector_type(8)))  float  v8f;

#define D_MODEL   2048
#define NUM_HEADS 16
#define HEAD_DIM  128
#define BATCH     2
#define SEQ       2048
#define MROWS     (BATCH * SEQ)      // 4096 rows for the dense GEMMs
#define QTILES    (SEQ / 16)         // 128 q-tiles per (b,h)
#define ATTN_WAVES 4

union BF16Frag {
    v16bf v;
    uint4 q[2];
};

// A-fragment (16x32, MxK): lane r (0..15), half h. Per ISA 7.12.2 the lane
// holds row r, K in {8h..8h+7} U {16+8h..16+8h+7} -> two 16B contiguous loads.
__device__ __forceinline__ v16bf load_a_frag(const __bf16* base, int ld) {
    int lane = threadIdx.x & 31;
    int r = lane & 15, h = lane >> 4;
    BF16Frag f;
    f.q[0] = *reinterpret_cast<const uint4*>(base + (size_t)r * ld + h * 8);
    f.q[1] = *reinterpret_cast<const uint4*>(base + (size_t)r * ld + 16 + h * 8);
    return f.v;
}

// B-fragment (32x16, KxN) loaded from B^T stored row-major ([N][K], ld in
// elements). Same K order as the A-fragment: lane half h holds
// K in {8h..8h+7} U {16+8h..16+8h+7}, lane = column N -> two 16B loads.
__device__ __forceinline__ v16bf load_b_frag(const __bf16* baseT, int ld) {
    int lane = threadIdx.x & 31;
    int r = lane & 15, h = lane >> 4;
    const __bf16* p = baseT + (size_t)r * ld;
    BF16Frag f;
    f.q[0] = *reinterpret_cast<const uint4*>(p + h * 8);
    f.q[1] = *reinterpret_cast<const uint4*>(p + 16 + h * 8);
    return f.v;
}

__device__ __forceinline__ v8f wmma_bf16(v16bf a, v16bf b, v8f c) {
    return __builtin_amdgcn_wmma_f32_16x16x32_bf16(
        /*neg_a=*/false, a, /*neg_b=*/false, b,
        /*c_mod=*/(short)0, c, /*reuse_a=*/false, /*reuse_b=*/false);
}

// ---------------------------------------------------------------------------
// Dense GEMM core: C[64x64] tile per wave, A [M x 2048] bf16 row-major,
// Bt [N x 2048] bf16 row-major (= B transposed). 16 WMMAs per K-step of 32.
// ---------------------------------------------------------------------------
__device__ __forceinline__ void gemm_acc_64x64(const __bf16* __restrict__ A,
                                               const __bf16* __restrict__ Bt,
                                               int m0, int n0, v8f acc[4][4]) {
    const v8f zero = {0.f, 0.f, 0.f, 0.f, 0.f, 0.f, 0.f, 0.f};
#pragma unroll
    for (int i = 0; i < 4; ++i)
#pragma unroll
        for (int j = 0; j < 4; ++j) acc[i][j] = zero;

    for (int kk = 0; kk < D_MODEL; kk += 32) {
        v16bf a[4], b[4];
#pragma unroll
        for (int i = 0; i < 4; ++i)
            a[i] = load_a_frag(A + (size_t)(m0 + 16 * i) * D_MODEL + kk, D_MODEL);
#pragma unroll
        for (int j = 0; j < 4; ++j)
            b[j] = load_b_frag(Bt + (size_t)(n0 + 16 * j) * D_MODEL + kk, D_MODEL);
#pragma unroll
        for (int i = 0; i < 4; ++i)
#pragma unroll
            for (int j = 0; j < 4; ++j)
                acc[i][j] = wmma_bf16(a[i], b[j], acc[i][j]);
    }
}

// 8 waves/block: 2 (M) x 4 (N) waves -> 128x256 block tile.
__global__ void __launch_bounds__(256) gemm_out_bf16(const __bf16* __restrict__ A,
                                                     const __bf16* __restrict__ Bt,
                                                     __bf16* __restrict__ C) {
    int wave = threadIdx.x >> 5, lane = threadIdx.x & 31;
    int m0 = blockIdx.y * 128 + (wave >> 2) * 64;
    int n0 = blockIdx.x * 256 + (wave & 3) * 64;
    v8f acc[4][4];
    gemm_acc_64x64(A, Bt, m0, n0, acc);
    int r = lane & 15, hh = lane >> 4;
#pragma unroll
    for (int i = 0; i < 4; ++i)
#pragma unroll
        for (int j = 0; j < 4; ++j)
#pragma unroll
            for (int e = 0; e < 8; ++e)
                C[(size_t)(m0 + 16 * i + e + 8 * hh) * D_MODEL + n0 + 16 * j + r] =
                    (__bf16)acc[i][j][e];
}

__global__ void __launch_bounds__(256) gemm_out_f32(const __bf16* __restrict__ A,
                                                    const __bf16* __restrict__ Bt,
                                                    float* __restrict__ C) {
    int wave = threadIdx.x >> 5, lane = threadIdx.x & 31;
    int m0 = blockIdx.y * 128 + (wave >> 2) * 64;
    int n0 = blockIdx.x * 256 + (wave & 3) * 64;
    v8f acc[4][4];
    gemm_acc_64x64(A, Bt, m0, n0, acc);
    int r = lane & 15, hh = lane >> 4;
#pragma unroll
    for (int i = 0; i < 4; ++i)
#pragma unroll
        for (int j = 0; j < 4; ++j)
#pragma unroll
            for (int e = 0; e < 8; ++e)
                C[(size_t)(m0 + 16 * i + e + 8 * hh) * D_MODEL + n0 + 16 * j + r] =
                    acc[i][j][e];
}

// V projection that writes its result pre-transposed: Vt[b][h][d][s].
__global__ void __launch_bounds__(256) gemm_out_vt(const __bf16* __restrict__ A,
                                                   const __bf16* __restrict__ Bt,
                                                   __bf16* __restrict__ Vt) {
    int wave = threadIdx.x >> 5, lane = threadIdx.x & 31;
    int m0 = blockIdx.y * 128 + (wave >> 2) * 64;
    int n0 = blockIdx.x * 256 + (wave & 3) * 64;
    v8f acc[4][4];
    gemm_acc_64x64(A, Bt, m0, n0, acc);
    int r = lane & 15, hh = lane >> 4;
#pragma unroll
    for (int i = 0; i < 4; ++i)
#pragma unroll
        for (int j = 0; j < 4; ++j)
#pragma unroll
            for (int e = 0; e < 8; ++e) {
                int row = m0 + 16 * i + e + 8 * hh;     // b*SEQ + s
                int col = n0 + 16 * j + r;              // h*128 + d
                int b = row >> 11, s = row & (SEQ - 1);
                int h = col >> 7,  d = col & (HEAD_DIM - 1);
                Vt[((size_t)(b * NUM_HEADS + h) * HEAD_DIM + d) * SEQ + s] =
                    (__bf16)acc[i][j][e];
            }
}

// ---------------------------------------------------------------------------
// Flash-attention: one 16-row Q tile per wave, 32 keys per inner step.
// Causal: loop bound k0 < q0+16 guarantees k0 <= q0 (k0 % 32 == 0, q0 % 16
// == 0), so every row has >=1 unmasked key per tile -> no -inf edge cases.
// ---------------------------------------------------------------------------
__global__ void __launch_bounds__(ATTN_WAVES * 32)
attn_fwd(const __bf16* __restrict__ Qm, const __bf16* __restrict__ Km,
         const __bf16* __restrict__ Vt, __bf16* __restrict__ Om) {
    __shared__ __align__(16) __bf16 plds[ATTN_WAVES][16 * 32];

    int wave = threadIdx.x >> 5, lane = threadIdx.x & 31;
    int r = lane & 15, hh = lane >> 4;
    int gw = blockIdx.x * ATTN_WAVES + wave;
    int qt = gw & (QTILES - 1);
    int h  = (gw >> 7) & (NUM_HEADS - 1);
    int b  = gw >> 11;
    int q0 = qt * 16;
    const size_t bS = (size_t)b * SEQ;

    const __bf16* qbase = Qm + (bS + q0) * D_MODEL + h * HEAD_DIM;
    v16bf qf[4];
#pragma unroll
    for (int t = 0; t < 4; ++t) qf[t] = load_a_frag(qbase + t * 32, D_MODEL);

    const v8f zero = {0.f, 0.f, 0.f, 0.f, 0.f, 0.f, 0.f, 0.f};
    v8f o[8];
#pragma unroll
    for (int t2 = 0; t2 < 8; ++t2) o[t2] = zero;
    float m[8], l[8];
#pragma unroll
    for (int j = 0; j < 8; ++j) { m[j] = -3.0e38f; l[j] = 0.f; }

    const float scale = 0.08838834764831845f;  // 1/sqrt(HEAD_DIM)
    const int kend = q0 + 16;

    for (int k0 = 0; k0 < kend; k0 += 32) {
        // --- scores: S = Q (16xd) @ K^T (d x 32 keys), f32 accumulate
        v8f s0 = zero, s1 = zero;
#pragma unroll
        for (int t = 0; t < 4; ++t) {
            const __bf16* kb0 = Km + (bS + k0) * D_MODEL + h * HEAD_DIM + t * 32;
            v16bf kf0 = load_b_frag(kb0, D_MODEL);
            v16bf kf1 = load_b_frag(kb0 + (size_t)16 * D_MODEL, D_MODEL);
            s0 = wmma_bf16(qf[t], kf0, s0);
            s1 = wmma_bf16(qf[t], kf1, s1);
        }
        // --- causal mask + online softmax (rows striped: row = j + 8*hh)
        float p0[8], p1[8];
#pragma unroll
        for (int j = 0; j < 8; ++j) {
            int row = q0 + j + 8 * hh;
            float v0 = s0[j] * scale; if (k0 + r      > row) v0 = -3.0e38f;
            float v1 = s1[j] * scale; if (k0 + 16 + r > row) v1 = -3.0e38f;
            float mx = fmaxf(v0, v1);
            mx = fmaxf(mx, __shfl_xor(mx, 1, 32));
            mx = fmaxf(mx, __shfl_xor(mx, 2, 32));
            mx = fmaxf(mx, __shfl_xor(mx, 4, 32));
            mx = fmaxf(mx, __shfl_xor(mx, 8, 32));
            float mn = fmaxf(m[j], mx);
            float alpha = __expf(m[j] - mn);
            m[j] = mn;
            float e0 = __expf(v0 - mn), e1 = __expf(v1 - mn);
            p0[j] = e0; p1[j] = e1;
            float sum = e0 + e1;
            sum += __shfl_xor(sum, 1, 32);
            sum += __shfl_xor(sum, 2, 32);
            sum += __shfl_xor(sum, 4, 32);
            sum += __shfl_xor(sum, 8, 32);
            l[j] = l[j] * alpha + sum;
#pragma unroll
            for (int t2 = 0; t2 < 8; ++t2) o[t2][j] *= alpha;
        }
        // --- P: C-layout -> A-fragment layout via wave-private LDS bounce
        __bf16* pl = plds[wave];
#pragma unroll
        for (int j = 0; j < 8; ++j) {
            pl[(j + 8 * hh) * 32 + r]      = (__bf16)p0[j];
            pl[(j + 8 * hh) * 32 + 16 + r] = (__bf16)p1[j];
        }
        asm volatile("s_wait_dscnt 0" ::: "memory");  // cross-lane LDS RAW
        v16bf pf = load_a_frag(pl, 32);
        // --- O += P (16x32 keys) @ V (32 keys x 128d), Vt is [b][h][d][s]
        const __bf16* vb = Vt + (size_t)(b * NUM_HEADS + h) * HEAD_DIM * SEQ + k0;
#pragma unroll
        for (int t2 = 0; t2 < 8; ++t2) {
            v16bf vf = load_b_frag(vb + (size_t)(t2 * 16) * SEQ, SEQ);
            o[t2] = wmma_bf16(pf, vf, o[t2]);
        }
    }

    float rl[8];
#pragma unroll
    for (int j = 0; j < 8; ++j) rl[j] = 1.f / l[j];
    __bf16* obase = Om + (bS + q0) * D_MODEL + h * HEAD_DIM;
#pragma unroll
    for (int t2 = 0; t2 < 8; ++t2)
#pragma unroll
        for (int j = 0; j < 8; ++j)
            obase[(size_t)(j + 8 * hh) * D_MODEL + t2 * 16 + r] =
                (__bf16)(o[t2][j] * rl[j]);
}

// ---------------------------------------------------------------------------
// Precision staging kernels
// ---------------------------------------------------------------------------
__global__ void cvt_f32_bf16(const float* __restrict__ x, __bf16* __restrict__ y, int n) {
    int i = blockIdx.x * blockDim.x + threadIdx.x;
    if (i < n) y[i] = (__bf16)x[i];
}

// w: [K=2048][N=2048] f32 row-major -> wt: [N][K] bf16 (so B-frags load contiguously)
__global__ void transpose_w_bf16(const float* __restrict__ w, __bf16* __restrict__ wt) {
    int i = blockIdx.x * blockDim.x + threadIdx.x;
    if (i >= D_MODEL * D_MODEL) return;
    int k = i & (D_MODEL - 1);
    int n = i >> 11;
    wt[(size_t)n * D_MODEL + k] = (__bf16)w[(size_t)k * D_MODEL + n];
}

// ---------------------------------------------------------------------------
extern "C" void kernel_launch(void* const* d_in, const int* in_sizes, int n_in,
                              void* d_out, int out_size, void* d_ws, size_t ws_size,
                              hipStream_t stream) {
    const float* x  = (const float*)d_in[0];
    // d_in[1] = additive causal mask; causality is applied analytically.
    const float* wq = (const float*)d_in[2];
    const float* wk = (const float*)d_in[3];
    const float* wv = (const float*)d_in[4];
    const float* wo = (const float*)d_in[5];
    float* out = (float*)d_out;

    char* ws = (char*)d_ws;                       // 112 MB used, all rewritten
    __bf16* xb  = (__bf16*)(ws);                  // 16 MB  [4096][2048]
    __bf16* wqt = (__bf16*)(ws + (16u  << 20));   // 8 MB   [N][K]
    __bf16* wkt = (__bf16*)(ws + (24u  << 20));
    __bf16* wvt = (__bf16*)(ws + (32u  << 20));
    __bf16* wot = (__bf16*)(ws + (40u  << 20));
    __bf16* qb  = (__bf16*)(ws + (48u  << 20));   // 16 MB  [4096][2048]
    __bf16* kb  = (__bf16*)(ws + (64u  << 20));   // 16 MB
    __bf16* vt  = (__bf16*)(ws + (80u  << 20));   // 16 MB  [B][H][128][SEQ]
    __bf16* att = (__bf16*)(ws + (96u  << 20));   // 16 MB  [4096][2048]

    // Stage to bf16
    cvt_f32_bf16<<<(MROWS * D_MODEL) / 256, 256, 0, stream>>>(x, xb, MROWS * D_MODEL);
    const int wgrid = (D_MODEL * D_MODEL) / 256;
    transpose_w_bf16<<<wgrid, 256, 0, stream>>>(wq, wqt);
    transpose_w_bf16<<<wgrid, 256, 0, stream>>>(wk, wkt);
    transpose_w_bf16<<<wgrid, 256, 0, stream>>>(wv, wvt);
    transpose_w_bf16<<<wgrid, 256, 0, stream>>>(wo, wot);

    // QKV projections (V written pre-transposed for the PV WMMA)
    dim3 ggrid(D_MODEL / 256, MROWS / 128);       // (8, 32)
    gemm_out_bf16<<<ggrid, 256, 0, stream>>>(xb, wqt, qb);
    gemm_out_bf16<<<ggrid, 256, 0, stream>>>(xb, wkt, kb);
    gemm_out_vt  <<<ggrid, 256, 0, stream>>>(xb, wvt, vt);

    // Causal flash attention
    attn_fwd<<<(BATCH * NUM_HEADS * QTILES) / ATTN_WAVES, ATTN_WAVES * 32, 0, stream>>>(
        qb, kb, vt, att);

    // Output projection, f32 result
    gemm_out_f32<<<ggrid, 256, 0, stream>>>(att, wot, out);
}
